// LSTM_Cell_19567871001103
// MI455X (gfx1250) — compile-verified, matrix operand order checked by reading, not run on hardware
//
#include <hip/hip_runtime.h>

// ---------------------------------------------------------------------------
// Fused LSTM cell for MI455X (gfx1250, wave32, WMMA).
//   gates = [x|h] @ [Wf|Wi|Wc|Wo] + b     (GEMM M=8192, K=2048, N=4*1024)
//   c' = sig(f)*c + sig(i)*tanh(g);  h' = sig(o)*tanh(c')
// Plan: one-time fp32->bf16 pre-pass into d_ws (z concat; W transposed to
// [gate][n][k]) so the GEMM hot loop stages pure bf16 16B copies -- via
// global_load_async_to_lds_b128 (ASYNCcnt path, builtin confirmed present).
// GEMM: bf16 WMMA, fp32 accum, 256-thread block (8 waves, 4Mx2N) owns a
// [128 x 32] output tile and all 4 gates; double-buffered LDS, BK=64.
// Falls back to convert-in-loop kernel when ws_size < 48MB.
// ---------------------------------------------------------------------------

typedef __attribute__((ext_vector_type(16))) __bf16 v16bf;
typedef __attribute__((ext_vector_type(8)))  __bf16 v8bf;
typedef __attribute__((ext_vector_type(8)))  float  v8f;
typedef unsigned short ushort_t;

#define B_M   8192
#define DX_K  1024
#define U_N   1024
#define K_TOT 2048          // DX + U
#define BM    128           // block tile rows (batch)
#define BN    32            // block tile cols (per gate)
#define BK    64            // K per stage round (2 WMMA steps)
#define LDSTR 72            // BK + 8 pad halves: 144B = 36-bank row stride ->
                            // 16 rows land 4 banks apart: 0-conflict b128

#define WS_Z_ELEMS   ((size_t)B_M * K_TOT)            // bf16 z     (32MB)
#define WS_W_ELEMS   ((size_t)4 * U_N * K_TOT)        // bf16 W^T   (16MB)
#define WS_NEEDED    ((WS_Z_ELEMS + WS_W_ELEMS) * sizeof(ushort_t))

#if __has_builtin(__builtin_amdgcn_global_load_async_to_lds_b128)
#define ASYNC_LDS 1
#else
#define ASYNC_LDS 0
#endif

// pack two f32 -> bf16x2 in one dword (lo = a, hi = b)
__device__ __forceinline__ unsigned pk_bf16(float a, float b) {
#if __has_builtin(__builtin_amdgcn_cvt_pk_bf16_f32)
    auto p = __builtin_amdgcn_cvt_pk_bf16_f32(a, b);
    unsigned r;
    __builtin_memcpy(&r, &p, 4);
    return r;
#else
    // round-half-up then take high halves with one v_perm_b32
    unsigned ra = __float_as_uint(a) + 0x8000u;
    unsigned rb = __float_as_uint(b) + 0x8000u;
    return __builtin_amdgcn_perm(rb, ra, 0x07060302u);
#endif
}

__device__ __forceinline__ float fast_sigmoid(float x) {
    return 1.0f / (1.0f + __expf(-x));
}
__device__ __forceinline__ float fast_tanh(float x) {
    float e = __expf(2.0f * x);
    return (e - 1.0f) / (e + 1.0f);
}

// 16B global->LDS copy: async DMA when available, reg bounce otherwise.
// Builtin prototype (from hipcc diagnostic): param1 = vector_size(16) int,
// __device__ (AS1) pointer.  Build exact-type pointers via int casts; the
// low 32 bits of a generic LDS pointer are the LDS byte offset (ISA aperture).
__device__ __forceinline__ void copy16(const ushort_t* g, ushort_t* l) {
#if ASYNC_LDS
    typedef int v4i_t __attribute__((vector_size(4 * sizeof(int))));
    typedef __attribute__((address_space(1))) v4i_t* gptr_t;
    typedef __attribute__((address_space(3))) v4i_t* lptr_t;
    __builtin_amdgcn_global_load_async_to_lds_b128(
        (gptr_t)(unsigned long long)g,
        (lptr_t)(unsigned)(unsigned long long)l,
        0, 0);
#else
    uint4 t = *(const uint4*)g;
    *(uint4*)l = t;
#endif
}

__device__ __forceinline__ void wait_async() {
#if ASYNC_LDS
#if __has_builtin(__builtin_amdgcn_s_wait_asynccnt)
    __builtin_amdgcn_s_wait_asynccnt(0);
#else
    asm volatile("s_wait_asynccnt 0" ::: "memory");
#endif
#endif
}

// One bf16 fragment per CDNA5 16-bit A/B layout: lane<16 holds K 0..7 & 16..23
// of its row/col, lane>=16 holds K 8..15 & 24..31 (two 16B-aligned chunks).
__device__ __forceinline__ v16bf load_frag(const ushort_t* base, int k0, int k1) {
    v8bf lo = *(const v8bf*)(base + k0);
    v8bf hi = *(const v8bf*)(base + k1);
    return __builtin_shufflevector(lo, hi, 0, 1, 2, 3, 4, 5, 6, 7,
                                   8, 9, 10, 11, 12, 13, 14, 15);
}

// ---------------------------------------------------------------------------
// Pre-pass 1: z = concat(x,h) -> bf16 [8192][2048]
// ---------------------------------------------------------------------------
__global__ __launch_bounds__(256)
void convert_z_kernel(const float* __restrict__ x, const float* __restrict__ h,
                      ushort_t* __restrict__ zb) {
    const size_t idx = (size_t)blockIdx.x * 256 + threadIdx.x; // 2M threads
    const int row   = (int)(idx >> 8);     // 256 8-elem chunks per row
    const int chunk = (int)(idx & 255);
    const float* __restrict__ src = (chunk < 128)
        ? (x + (size_t)row * 1024 + chunk * 8)
        : (h + (size_t)row * 1024 + (chunk - 128) * 8);
    const float4 v0 = ((const float4*)src)[0];
    const float4 v1 = ((const float4*)src)[1];
    uint4 o;
    o.x = pk_bf16(v0.x, v0.y);
    o.y = pk_bf16(v0.z, v0.w);
    o.z = pk_bf16(v1.x, v1.y);
    o.w = pk_bf16(v1.z, v1.w);
    *(uint4*)(zb + (size_t)row * K_TOT + chunk * 8) = o;
}

// ---------------------------------------------------------------------------
// Pre-pass 2: W^T -> bf16 [gate][n (1024)][k (2048)]  (32x32 LDS transpose)
// ---------------------------------------------------------------------------
__global__ __launch_bounds__(256)
void convert_w_kernel(const float* __restrict__ Wf, const float* __restrict__ Wi,
                      const float* __restrict__ Wc, const float* __restrict__ Wo,
                      ushort_t* __restrict__ Wt) {
    __shared__ float tile[32][33];
    const int g = blockIdx.z;
    const float* __restrict__ W = (g == 0) ? Wf : (g == 1) ? Wi : (g == 2) ? Wc : Wo;
    const int k0 = blockIdx.x * 32;        // 64 blocks over K
    const int n0 = blockIdx.y * 32;        // 32 blocks over N
    const int r  = threadIdx.x >> 3;       // 0..31 (k within tile)
    const int c  = threadIdx.x & 7;        // float4 col
    const float4 v = *(const float4*)(W + (size_t)(k0 + r) * U_N + n0 + c * 4);
    tile[r][c * 4 + 0] = v.x;
    tile[r][c * 4 + 1] = v.y;
    tile[r][c * 4 + 2] = v.z;
    tile[r][c * 4 + 3] = v.w;
    __syncthreads();
    #pragma unroll
    for (int s = threadIdx.x; s < 512; s += 256) {     // 512 k-pair dwords
        const int n  = s >> 4;                          // 0..31
        const int kp = s & 15;                          // 0..15
        *(unsigned*)(Wt + (size_t)g * U_N * K_TOT +
                     (size_t)(n0 + n) * K_TOT + k0 + 2 * kp) =
            pk_bf16(tile[2 * kp][n], tile[2 * kp + 1][n]);
    }
}

// ---------------------------------------------------------------------------
// Main GEMM + gating.  PRE=true: bf16 operands from d_ws (async staging).
//                      PRE=false: fp32 operands, convert during staging.
// ---------------------------------------------------------------------------
template <bool PRE>
__global__ __launch_bounds__(256, 2)
void lstm_cell_wmma_kernel(const ushort_t* __restrict__ zb,
                           const ushort_t* __restrict__ Wt,
                           const float* __restrict__ x,
                           const float* __restrict__ h,
                           const float* __restrict__ cIn,
                           const float* __restrict__ Wf,
                           const float* __restrict__ bf_,
                           const float* __restrict__ Wi,
                           const float* __restrict__ bi_,
                           const float* __restrict__ Wc,
                           const float* __restrict__ bc_,
                           const float* __restrict__ Wo,
                           const float* __restrict__ bo_,
                           float* __restrict__ out) {
    __shared__ ushort_t ldsA[2][BM * LDSTR];      // [buf][m][k]       2x18KB
    __shared__ ushort_t ldsB[2][4][BN * LDSTR];   // [buf][gate][n][k] 2x18KB

    const int tid    = threadIdx.x;
    const int lane   = tid & 31;
    const int waveId = tid >> 5;
    const int wm     = waveId >> 1;     // 0..3 -> M offset wm*32
    const int wn     = waveId & 1;      // 0..1 -> N offset wn*16
    const int half   = (lane >> 4) & 1;
    const int lrow   = lane & 15;
    const int k0     = half * 8;
    const int k1     = 16 + half * 8;

    const int mBlock = blockIdx.x * BM;
    const int nBlock = blockIdx.y * BN;

    v8f acc[2][4] = {};                 // [mTile][gate] fp32 accumulators

    // ---- PRE staging: 2048 16B chunks/round, 8 per thread ---------------
    auto stagePre = [&](int buf, int kt) {
        #pragma unroll
        for (int i = 0; i < 4; ++i) {                  // A: 1024 chunks
            const int id  = tid + 256 * i;
            const int row = id >> 3, c = id & 7;
            copy16(zb + (size_t)(mBlock + row) * K_TOT + kt + c * 8,
                   &ldsA[buf][row * LDSTR + c * 8]);
        }
        #pragma unroll
        for (int i = 0; i < 4; ++i) {                  // B: 1024 chunks
            const int id = tid + 256 * i;
            const int g  = id >> 8;
            const int n  = (id & 255) >> 3, c = id & 7;
            copy16(Wt + (size_t)g * U_N * K_TOT +
                       (size_t)(nBlock + n) * K_TOT + kt + c * 8,
                   &ldsB[buf][g][n * LDSTR + c * 8]);
        }
    };

    // ---- non-PRE staging state (convert during staging) -----------------
    const float* __restrict__ Wg[4] = {Wf, Wi, Wc, Wo};
    const int cA = tid & 15, rA = tid >> 4;     // A: 16 f4-cols x 16 rows
    const int cB = tid & 7,  kp = tid >> 3;     // B: 8 f4-cols x 32 k-pairs
    float4 ra[8];
    float4 rb0[4], rb1[4];

    auto loadG = [&](int kt) {
        const float* __restrict__ srcA = (kt < DX_K) ? (x + kt) : (h + (kt - DX_K));
        #pragma unroll
        for (int p = 0; p < 8; ++p)
            ra[p] = *(const float4*)(srcA +
                (size_t)(mBlock + p * 16 + rA) * 1024 + cA * 4);
        #pragma unroll
        for (int g = 0; g < 4; ++g) {
            const float* __restrict__ w =
                Wg[g] + (size_t)(kt + 2 * kp) * U_N + nBlock + cB * 4;
            rb0[g] = *(const float4*)(w);
            rb1[g] = *(const float4*)(w + U_N);
        }
    };
    auto storeL = [&](int buf) {
        #pragma unroll
        for (int p = 0; p < 8; ++p) {
            uint2 pk;
            pk.x = pk_bf16(ra[p].x, ra[p].y);
            pk.y = pk_bf16(ra[p].z, ra[p].w);
            *(uint2*)&ldsA[buf][(p * 16 + rA) * LDSTR + cA * 4] = pk;
        }
        #pragma unroll
        for (int g = 0; g < 4; ++g) {
            const float* f0 = &rb0[g].x;
            const float* f1 = &rb1[g].x;
            #pragma unroll
            for (int j = 0; j < 4; ++j)
                *(unsigned*)&ldsB[buf][g][(cB * 4 + j) * LDSTR + 2 * kp] =
                    pk_bf16(f0[j], f1[j]);
        }
    };

    // ---- 16 WMMAs per round (2 K-steps x 2 M-tiles x 4 gates) -----------
    auto compute = [&](int buf) {
        #pragma unroll
        for (int ks = 0; ks < 2; ++ks) {
            const v16bf a0 = load_frag(
                &ldsA[buf][(wm * 32 +  0 + lrow) * LDSTR + ks * 32], k0, k1);
            const v16bf a1 = load_frag(
                &ldsA[buf][(wm * 32 + 16 + lrow) * LDSTR + ks * 32], k0, k1);
            #pragma unroll
            for (int g = 0; g < 4; ++g) {
                const v16bf b = load_frag(
                    &ldsB[buf][g][(wn * 16 + lrow) * LDSTR + ks * 32], k0, k1);
                acc[0][g] = __builtin_amdgcn_wmma_f32_16x16x32_bf16(
                    false, a0, false, b, (short)0, acc[0][g], false, false);
                acc[1][g] = __builtin_amdgcn_wmma_f32_16x16x32_bf16(
                    false, a1, false, b, (short)0, acc[1][g], false, false);
            }
        }
    };

    // ---- software-pipelined main loop -----------------------------------
    int buf = 0;
    if constexpr (PRE) {
        stagePre(0, 0);
        for (int kt = 0; kt < K_TOT; kt += BK) {
            wait_async();                      // our async writes to LDS done
            __syncthreads();                   // everyone's writes visible
            const bool more = (kt + BK) < K_TOT;
            if (more) stagePre(buf ^ 1, kt + BK);  // DMA overlaps WMMAs
            compute(buf);
            buf ^= 1;
        }
    } else {
        loadG(0);
        storeL(0);
        for (int kt = 0; kt < K_TOT; kt += BK) {
            __syncthreads();
            const bool more = (kt + BK) < K_TOT;
            if (more) loadG(kt + BK);
            compute(buf);
            if (more) storeL(buf ^ 1);
            buf ^= 1;
        }
    }

    // ---- fused gating epilogue ------------------------------------------
    // C/D layout: VGPR r of lane l holds element (M = r + 8*(l>=16), N = l&15).
    const int u  = nBlock + wn * 16 + lrow;
    const float Bf = bf_[u], Bi = bi_[u], Bc = bc_[u], Bo = bo_[u];
    const int mWave = mBlock + wm * 32;

    #pragma unroll
    for (int mt = 0; mt < 2; ++mt) {
        #pragma unroll
        for (int r = 0; r < 8; ++r) {
            const int row = mWave + mt * 16 + r + half * 8;
            const float fg = fast_sigmoid(acc[mt][0][r] + Bf);
            const float ig = fast_sigmoid(acc[mt][1][r] + Bi);
            const float gg = fast_tanh   (acc[mt][2][r] + Bc);
            const float og = fast_sigmoid(acc[mt][3][r] + Bo);
            const float cOld = cIn[(size_t)row * U_N + u];
            const float cNew = fg * cOld + ig * gg;
            const float hNew = og * fast_tanh(cNew);
            out[(size_t)row * U_N + u]                     = hNew;  // h_new
            out[(size_t)B_M * U_N + (size_t)row * U_N + u] = cNew;  // c_new
        }
    }
}

extern "C" void kernel_launch(void* const* d_in, const int* in_sizes, int n_in,
                              void* d_out, int out_size, void* d_ws, size_t ws_size,
                              hipStream_t stream) {
    (void)in_sizes; (void)n_in; (void)out_size;
    const float* x   = (const float*)d_in[0];
    const float* h   = (const float*)d_in[1];
    const float* c   = (const float*)d_in[2];
    const float* Wf  = (const float*)d_in[3];
    const float* bf_ = (const float*)d_in[4];
    const float* Wi  = (const float*)d_in[5];
    const float* bi_ = (const float*)d_in[6];
    const float* Wc  = (const float*)d_in[7];
    const float* bc_ = (const float*)d_in[8];
    const float* Wo  = (const float*)d_in[9];
    const float* bo_ = (const float*)d_in[10];
    float* out = (float*)d_out;

    dim3 grid(B_M / BM, U_N / BN);   // 64 x 32 = 2048 blocks
    dim3 block(256);                  // 8 waves (wave32)

    if (ws_size >= WS_NEEDED) {
        ushort_t* zb = (ushort_t*)d_ws;
        ushort_t* Wt = zb + WS_Z_ELEMS;
        // pre-pass: fp32 -> bf16 (z concat; W transposed per gate)
        convert_z_kernel<<<dim3((unsigned)(WS_Z_ELEMS / 8 / 256)), block, 0, stream>>>(x, h, zb);
        convert_w_kernel<<<dim3(K_TOT / 32, U_N / 32, 4), block, 0, stream>>>(Wf, Wi, Wc, Wo, Wt);
        lstm_cell_wmma_kernel<true><<<grid, block, 0, stream>>>(
            zb, Wt, x, h, c, Wf, bf_, Wi, bi_, Wc, bc_, Wo, bo_, out);
    } else {
        lstm_cell_wmma_kernel<false><<<grid, block, 0, stream>>>(
            nullptr, nullptr, x, h, c, Wf, bf_, Wi, bi_, Wc, bc_, Wo, bo_, out);
    }
}
